// GraphAttentionLayer_64596308132447
// MI455X (gfx1250) — matrix operand order repaired and checked by reading, compiled, not run on hardware
//
#include <hip/hip_runtime.h>
#include <hip/hip_bf16.h>
#include <math.h>

// ---------------------------------------------------------------------------
// Fused GAT layer for MI455X (gfx1250, wave32, WMMA).
//   N=64, C=64, T=300, V=25, F=64.  One workgroup = one (n, 10-t block).
//   GEMM1 (x^T W) and GEMM2 (att x h) via v_wmma_f32_16x16x32_f16, f32 accum.
//   Row index m = t*32 + v (v 25..31 zero padded): power-of-two strides,
//   branch-free epilogues, packed b128 LDS stores, GEMM2 K=32 pad = free.
//   Output is staged through LDS so global writes are 64B-contiguous rows.
//   ~128 KB LDS -> 2 workgroups per WGP.
// ---------------------------------------------------------------------------

typedef __attribute__((ext_vector_type(16))) _Float16 v16h;
typedef __attribute__((ext_vector_type(8)))  _Float16 v8h;   // one 16B chunk
typedef __attribute__((ext_vector_type(8)))  float    v8f;

#define Nn 64
#define Cc 64
#define Tt 300
#define Vv 25
#define Ff 64

#define TB    10           // t's per workgroup (300/10 = 30 blocks per n)
#define Mp    320          // TB*32 rows, m = t*32 + v, 20 WMMA M-tiles
#define AStr  72           // Ah row stride in halves (144B = 9*16B)
#define MStr  328          // Ht row stride in halves (656B: 16B-aligned, bank-spread)
#define JStr  40           // Att row stride in halves (80B = 5*16B)
#define HStr  67           // stage row stride in dwords (odd -> conflict-free columns)
#define HPW   (16*HStr + 4)  // per-wave stage floats (16B-aligned chunk)

#define ALPHA   0.2f
#define NEG_INF -9000000000000000.0f

union AFrag { v16h v; v8h h[2]; };

__global__ __launch_bounds__(256)
void gat_fused_kernel(const float* __restrict__ x,
                      const int*   __restrict__ adj,
                      const float* __restrict__ W,
                      const float* __restrict__ a,
                      float*       __restrict__ out)
{
    // ---- LDS staging (~128 KB).  Ah/Wt (phases 0-2) overlay the phase-5
    //      per-wave f32 output stage buffers (Ah is dead after GEMM1). ----
    __shared__ union {
        struct { alignas(16) _Float16 Ah[Mp*AStr];     // x tile [m][c]   46.1KB
                 alignas(16) _Float16 Wt[Ff*AStr]; } g; // W^T  [f][c]      9.2KB
        struct { alignas(16) float HP[8*HPW]; } s;      // out stage       34.5KB
    } OV;
    __shared__ alignas(16) _Float16 Ht [Ff*MStr];     // h^T [f][m]       42.0KB
    __shared__ alignas(16) _Float16 Att[TB*32*JStr];  // att [t][i][j]    25.6KB
    __shared__ float F1[Mp], F2[Mp];
    __shared__ float Avec[2*Ff];
    __shared__ int   AdjS[Vv*Vv];

    _Float16* Ah = OV.g.Ah;
    _Float16* Wt = OV.g.Wt;

    const int tid   = threadIdx.x;
    const int wave  = tid >> 5;         // 8 waves
    const int lane  = tid & 31;         // wave32
    const int L15   = lane & 15;
    const int khalf = lane >> 4;

    const int n  = blockIdx.y;
    const int t0 = blockIdx.x * TB;

    // ---------------- Phase 0: constants + Att zero (pads i>=25, j>=25) -------------
    {
        uint32_t* za = (uint32_t*)Att;
        for (int i = tid; i < (TB*32*JStr)/2; i += 256) za[i] = 0u;

        for (int i = tid; i < Cc*Ff; i += 256) {       // W^T -> LDS (f16)
            int c = i >> 6, f = i & 63;
            Wt[f*AStr + c] = (_Float16)W[i];
        }
        if (tid < 2*Ff) Avec[tid] = a[tid];
        for (int i = tid; i < Vv*Vv; i += 256) AdjS[i] = adj[i];
    }

    // ---------------- Phase 1: x tile -> Ah (f16, [m][c]); rows v>=25 zeroed --------
    // x[n,c,t,v]: for fixed (n,c) the 10x25 tile is 250 contiguous floats.
    {
        const size_t xb = (size_t)n * Cc * Tt * Vv + (size_t)t0 * Vv;
        for (int m = tid; m < Mp; m += 256) {
            const int t = m >> 5, v = m & 31;
            const bool valid = (v < Vv);
            // clamp address so every load is in-bounds; select pad zeros after
            const float* xr = x + xb + (size_t)t * Vv + (valid ? v : 0);
            for (int c = 0; c < Cc; c += 4) {
                if (c + 8 < Cc)
                    __builtin_prefetch(&xr[(size_t)(c + 8) * (Tt*Vv)], 0, 0);
                union { _Float16 h[4]; uint2 u2; } pk;
                #pragma unroll
                for (int q = 0; q < 4; ++q) {
                    const float val = xr[(size_t)(c + q) * (Tt*Vv)];
                    pk.h[q] = (_Float16)(valid ? val : 0.0f);
                }
                *(uint2*)&Ah[m*AStr + c] = pk.u2;      // packed b64 store
            }
        }
    }
    __syncthreads();

    // ---------------- Phase 2: GEMM1  h[m,f] = sum_c Ah[m,c]*W[c,f] -----------------
    // 20 M-tiles x 4 N-tiles; wave w owns mt = w, w+8, w+16.
    {
        AFrag bf[2][4];                                // hoisted W^T frags
        #pragma unroll
        for (int ks = 0; ks < 2; ++ks)
            #pragma unroll
            for (int nt = 0; nt < 4; ++nt) {
                const int bbase = (nt*16 + L15)*AStr + ks*32 + khalf*8;
                bf[ks][nt].h[0] = *(const v8h*)&Wt[bbase];
                bf[ks][nt].h[1] = *(const v8h*)&Wt[bbase + 16];
            }

        for (int mt = wave; mt < Mp/16; mt += 8) {
            v8f acc[4];
            #pragma unroll
            for (int nt = 0; nt < 4; ++nt)
                #pragma unroll
                for (int k = 0; k < 8; ++k) acc[nt][k] = 0.0f;

            #pragma unroll
            for (int ks = 0; ks < 2; ++ks) {           // K = 64 = 2 x 32
                AFrag af;                              // A: lane=M%16, khalf K-chunks
                const int abase = (mt*16 + L15)*AStr + ks*32 + khalf*8;
                af.h[0] = *(const v8h*)&Ah[abase];
                af.h[1] = *(const v8h*)&Ah[abase + 16];
                #pragma unroll
                for (int nt = 0; nt < 4; ++nt)
                    acc[nt] = __builtin_amdgcn_wmma_f32_16x16x32_f16(
                        false, af.v, false, bf[ks][nt].v, (short)0, acc[nt], false, false);
            }
            // Epilogue: rows m = mt*16 + 8*khalf + (0..7) contiguous in Ht[f][m]
            // -> 8 packed cvt + ONE ds_store_b128 per fragment.  Pad rows = 0.
            const int mp0 = mt*16 + 8*khalf;
            #pragma unroll
            for (int nt = 0; nt < 4; ++nt) {
                v8h pk;
                #pragma unroll
                for (int r = 0; r < 8; ++r) pk[r] = (_Float16)acc[nt][r];
                *(v8h*)&Ht[(nt*16 + L15)*MStr + mp0] = pk;
            }
        }
    }
    __syncthreads();

    // ---------------- Phase 3: f1 = h.a1, f2 = h.a2 (from Ht columns) ---------------
    if (tid < TB*Vv) {
        const int tl = tid / 25, j = tid - tl*25;
        const int m  = tl*32 + j;
        float s1 = 0.0f, s2 = 0.0f;
        #pragma unroll 8
        for (int f = 0; f < Ff; ++f) {
            const float hv = (float)Ht[f*MStr + m];
            s1 += hv * Avec[f];
            s2 += hv * Avec[Ff + f];
        }
        F1[tid] = s1; F2[tid] = s2;
    }
    __syncthreads();

    // ---------------- Phase 4: masked softmax rows (reference semantics) ------------
    if (tid < TB*Vv) {
        const int t = tid / 25, i = tid - t*25;
        const float f1v = F1[tid];
        float pbuf[Vv];
        float mx = NEG_INF;
        #pragma unroll
        for (int j = 0; j < Vv; ++j) {
            float e = f1v + F2[t*25 + j];
            e = (e > 0.0f) ? e : ALPHA * e;            // leaky_relu BEFORE mask
            if (AdjS[i*25 + j] <= 0) e = NEG_INF;
            pbuf[j] = e;
            mx = fmaxf(mx, e);
        }
        float sum = 0.0f;
        #pragma unroll
        for (int j = 0; j < Vv; ++j) {
            const float p = expf(pbuf[j] - mx);        // all-masked row -> uniform
            pbuf[j] = p;
            sum += p;
        }
        const float inv = 1.0f / sum;
        union { _Float16 h[26]; uint32_t u[13]; } row;
        #pragma unroll
        for (int j = 0; j < Vv; ++j) row.h[j] = (_Float16)(pbuf[j] * inv);
        row.h[25] = (_Float16)0.0f;                    // pad slot stays zero
        uint32_t* dst = (uint32_t*)&Att[(t*32 + i)*JStr];
        #pragma unroll
        for (int q = 0; q < 13; ++q) dst[q] = row.u[q];
    }
    __syncthreads();

    // ---------------- Phase 5: GEMM2 + ELU; stage 16x64 tiles in LDS so global
    //                  stores are 64B-contiguous runs of i per (f,t) row. -----------
    {
        float* HPw = OV.s.HP + wave * HPW;             // wave-private stage
        for (int t = wave; t < TB; t += 8) {
            const int tg = t0 + t;
            AFrag bf2[4];                              // hoisted h^T frags for this t
            #pragma unroll
            for (int nt = 0; nt < 4; ++nt) {
                const int bbase = (nt*16 + L15)*MStr + t*32 + khalf*8;
                bf2[nt].h[0] = *(const v8h*)&Ht[bbase];
                bf2[nt].h[1] = *(const v8h*)&Ht[bbase + 16];
            }
            #pragma unroll
            for (int mt = 0; mt < 2; ++mt) {
                AFrag af;
                const int abase = (t*32 + mt*16 + L15)*JStr + khalf*8;
                af.h[0] = *(const v8h*)&Att[abase];
                af.h[1] = *(const v8h*)&Att[abase + 16];
                #pragma unroll
                for (int nt = 0; nt < 4; ++nt) {
                    v8f d;
                    #pragma unroll
                    for (int k = 0; k < 8; ++k) d[k] = 0.0f;
                    d = __builtin_amdgcn_wmma_f32_16x16x32_f16(
                        false, af.v, false, bf2[nt].v, (short)0, d, false, false);
                    // ELU, then stage: row = local i (contiguous f per half-wave)
                    #pragma unroll
                    for (int r = 0; r < 8; ++r) {
                        float v = d[r];
                        v = (v > 0.0f) ? v : expm1f(v);
                        HPw[(r + 8*khalf)*HStr + nt*16 + L15] = v;
                    }
                }
                // same-wave LDS ops are in-order; make the RAW explicit
                asm volatile("s_wait_dscnt 0x0" ::: "memory");
                // transposed drain: half-wave writes 16 consecutive i (64B runs)
                #pragma unroll 4
                for (int fb = 0; fb < 32; ++fb) {
                    const int f  = fb*2 + khalf;
                    const int i2 = mt*16 + L15;
                    const float v = HPw[L15*HStr + f];
                    if (mt == 0 || i2 < Vv)
                        out[((size_t)(n*Ff + f)*Tt + tg)*Vv + i2] = v;
                }
            }
        }
    }
}

extern "C" void kernel_launch(void* const* d_in, const int* in_sizes, int n_in,
                              void* d_out, int out_size, void* d_ws, size_t ws_size,
                              hipStream_t stream) {
    const float* x   = (const float*)d_in[0];
    const int*   adj = (const int*)  d_in[1];
    const float* W   = (const float*)d_in[2];
    const float* a   = (const float*)d_in[3];
    float*       out = (float*)d_out;

    dim3 grid(Tt / TB, Nn, 1);   // 30 x 64 workgroups
    dim3 block(256, 1, 1);       // 8 wave32 waves
    gat_fused_kernel<<<grid, block, 0, stream>>>(x, adj, W, a, out);
}
